// EquivMultiHeadQueryModelOld_4114578670369
// MI455X (gfx1250) — matrix operand order repaired and checked by reading, not compiled, not run on hardware
//
#include <hip/hip_runtime.h>
#include <math.h>

#define NN    2048
#define DD    86
#define DLIN  172
#define SS    16
#define NB_   10
#define NQ_   16
#define MAXR  1.0f
#define FCUT  2.0f

typedef __attribute__((ext_vector_type(16))) __bf16 v16bf;
typedef __attribute__((ext_vector_type(8)))  float  v8f;

// ---------------- device helpers ----------------

__device__ __forceinline__ void sphf(float x, float y, float z, float* o) {
  const float s3 = 1.7320508f;
  o[0] = 1.f; o[1] = x; o[2] = y; o[3] = z;
  o[4] = s3 * x * y;
  o[5] = s3 * y * z;
  o[6] = 0.5f * (3.f * z * z - 1.f);
  o[7] = s3 * z * x;
  o[8] = 0.5f * s3 * (x * x - y * y);
  const float a = 1.0458f, b = 5.1235f, c = 0.8101f, d = 0.6614f, e = 2.5617f;
  const float inv = 0.75592895f; // 1/sqrt(1.75)
  o[9]  = a * y * (3.f * x * x - y * y) * inv;
  o[10] = b * x * y * z * inv;
  o[11] = c * y * (5.f * z * z - 1.f) * inv;
  o[12] = d * z * (5.f * z * z - 3.f) * inv;
  o[13] = c * x * (5.f * z * z - 1.f) * inv;
  o[14] = e * z * (x * x - y * y) * inv;
  o[15] = a * x * (x * x - 3.f * y * y) * inv;
}

__device__ __forceinline__ float gelu_tanh(float v) {
  float v3 = v * v * v;
  return 0.5f * v * (1.f + tanhf(0.7978845608f * (v + 0.044715f * v3)));
}

__device__ __forceinline__ void atomicMaxFloat(float* addr, float v) {
  if (v >= 0.f) atomicMax((int*)addr, __float_as_int(v));
  else          atomicMin((unsigned int*)addr, (unsigned int)__float_as_int(v));
}

// ---------------- edge geometry: sh[E,16], rad[E,10] ----------------

__global__ void edge_geom(const float* __restrict__ pos,
                          const int* __restrict__ src, const int* __restrict__ dst,
                          float* __restrict__ esh, float* __restrict__ erad, int E) {
  int e = blockIdx.x * 256 + threadIdx.x;
  if (e >= E) return;
  int s = src[e], d0 = dst[e];
  float dx = pos[d0 * 3 + 0] - pos[s * 3 + 0];
  float dy = pos[d0 * 3 + 1] - pos[s * 3 + 1];
  float dz = pos[d0 * 3 + 2] - pos[s * 3 + 2];
  float r = sqrtf(dx * dx + dy * dy + dz * dz + 1e-12f);
  float ir = 1.f / r;
  sphf(dx * ir, dy * ir, dz * ir, esh + (size_t)e * 16);
  float t = r > 1.f ? 1.f : r;
  float env = 0.5f * (cosf(3.14159265f * t) + 1.f);
#pragma unroll
  for (int b = 0; b < NB_; ++b) {
    float mu = (float)b * (1.f / 9.f);
    float zz = (r - mu) * 10.f;        // w = MAXR/NB = 0.1
    erad[(size_t)e * NB_ + b] = expf(-zz * zz) * env;
  }
}

// ---------------- weight convert/transpose to bf16 [96][Kpad] ----------------

__global__ void conv_w_bf16(const float* __restrict__ W, __bf16* __restrict__ Wt,
                            int Fin, int Kpad) {
  int idx = blockIdx.x * 256 + threadIdx.x;
  int total = 96 * Kpad;
  if (idx >= total) return;
  int d = idx / Kpad, k = idx - d * Kpad;
  float v = 0.f;
  if (d < DD && k < Fin * 16) v = W[(size_t)k * DD + d];   // W[f][s][d], k=f*16+s
  Wt[(size_t)d * Kpad + k] = (__bf16)v;
}

// ---------------- WMMA edge tensor-product GEMM ----------------
// out[e,d] = (sum_{f,s} xs[e,f]*sh[e,s]*W[f,s,d]) * (rad[e,:]@R[:,d])
// A[e, f*16+s] built on the fly (bf16), B = Wt bf16 [d][k], f32 accumulate.
// xs LDS tile is fully zero-padded to 88 columns so the K loop is branch-free.

__global__ __launch_bounds__(128)
void edge_gemm_wmma(const float* __restrict__ xin, int Fin,
                    const int* __restrict__ src,
                    const float* __restrict__ esh,
                    const float* __restrict__ erad,
                    const __bf16* __restrict__ Wt, int Kpad,
                    const float* __restrict__ R,   // [10, DD]
                    float* __restrict__ out,       // [E, 96]
                    int E) {
  __shared__ float xsL[64 * 88];
  __shared__ float radL[64 * 10];
  __shared__ float Rt[10 * 16];
  const int tid = threadIdx.x;
  const int wave = tid >> 5;
  const int lane = tid & 31;
  const int laneHalf = lane >> 4;
  const int ln16 = lane & 15;
  const int eBase = blockIdx.x * 64;
  const int dtile = blockIdx.y * 16;

  // stage xs: cover the whole padded tile, zero where invalid (no races, no
  // inner-loop guards needed later)
  for (int idx = tid; idx < 64 * 88; idx += 128) {
    int el = idx / 88, f = idx - el * 88;
    int eg = eBase + el;
    float v = 0.f;
    if (eg < E && f < Fin) v = xin[(size_t)src[eg] * Fin + f];
    xsL[idx] = v;
  }
  for (int idx = tid; idx < 64 * 10; idx += 128) {
    int el = idx / 10;
    int eg = eBase + el;
    radL[idx] = (eg < E) ? erad[(size_t)eg * 10 + (idx - el * 10)] : 0.f;
  }
  for (int idx = tid; idx < 160; idx += 128) {
    int b = idx >> 4, j = idx & 15;
    int d = dtile + j;
    Rt[idx] = (d < DD) ? R[b * DD + d] : 0.f;
  }
  __syncthreads();

  const int elocal = wave * 16 + ln16;
  const int eg = eBase + elocal;
  float shreg[16];
  if (eg < E) {
    const float4* sp = (const float4*)(esh + (size_t)eg * 16);
    float4 s0 = sp[0], s1 = sp[1], s2 = sp[2], s3 = sp[3];
    shreg[0] = s0.x; shreg[1] = s0.y; shreg[2]  = s0.z; shreg[3]  = s0.w;
    shreg[4] = s1.x; shreg[5] = s1.y; shreg[6]  = s1.z; shreg[7]  = s1.w;
    shreg[8] = s2.x; shreg[9] = s2.y; shreg[10] = s2.z; shreg[11] = s2.w;
    shreg[12] = s3.x; shreg[13] = s3.y; shreg[14] = s3.z; shreg[15] = s3.w;
  } else {
#pragma unroll
    for (int s = 0; s < 16; ++s) shreg[s] = 0.f;
  }

  // Pre-apply the laneHalf permutation of the ISA 16-bit A layout:
  // K = i + (i>=8 ? 8 : 0) + 8*laneHalf ; s = K & 15, f = f0 + (K>=16)
  float shsel[16];
#pragma unroll
  for (int i = 0; i < 16; ++i) {
    const int sA = (i < 8) ? i : (i - 8);      // laneHalf == 0
    const int sB = (i < 8) ? (i + 8) : i;      // laneHalf == 1
    shsel[i] = laneHalf ? shreg[sB] : shreg[sA];
  }

  const float* xrow = &xsL[elocal * 88];
  const __bf16* brow = Wt + (size_t)(dtile + ln16) * Kpad + 8 * laneHalf;

  v8f c = {0.f, 0.f, 0.f, 0.f, 0.f, 0.f, 0.f, 0.f};
  const int nT = Kpad >> 5;
  for (int t = 0; t < nT; ++t) {
    float2 xf = *(const float2*)(xrow + t * 2);   // branch-free: zero-padded
    v16bf a;
#pragma unroll
    for (int i = 0; i < 16; ++i)
      a[i] = (__bf16)(((i >= 8) ? xf.y : xf.x) * shsel[i]);
    v16bf b;
#pragma unroll
    for (int i = 0; i < 8; ++i) {
      b[i] = brow[t * 32 + i];
      b[8 + i] = brow[t * 32 + 16 + i];
    }
    __builtin_prefetch(brow + t * 32 + 32, 0, 3);
    c = __builtin_amdgcn_wmma_f32_16x16x32_bf16(false, a, false, b, (short)0, c,
                                                false, false);
  }

  const int dcol = dtile + ln16;
#pragma unroll
  for (int j = 0; j < 8; ++j) {
    int el2 = wave * 16 + j + 8 * laneHalf;    // C/D layout: M = j + 8*laneHalf
    int eg2 = eBase + el2;
    if (eg2 < E && dcol < DD) {
      float rr = 0.f;
#pragma unroll
      for (int b = 0; b < 10; ++b) rr += radL[el2 * 10 + b] * Rt[b * 16 + ln16];
      out[(size_t)eg2 * 96 + dcol] = c[j] * rr;
    }
  }
}

// ---------------- plain row-major GEMM (+optional bias) ----------------

__global__ void gemm_rm(const float* __restrict__ A, const float* __restrict__ B,
                        const float* __restrict__ bias, float* __restrict__ C,
                        int Nr, int K, int M) {
  int idx = blockIdx.x * 256 + threadIdx.x;
  if (idx >= Nr * M) return;
  int n = idx / M, m = idx - n * M;
  float s = bias ? bias[m] : 0.f;
  const float* a = A + (size_t)n * K;
  const float* b = B + m;
  for (int k = 0; k < K; ++k) s += a[k] * b[(size_t)k * M];
  C[idx] = s;
}

// ---------------- attention ----------------

__global__ void attn_init(float* mmax, float* den, float* agg) {
  int i = blockIdx.x * 256 + threadIdx.x;
  if (i < NN) { mmax[i] = -__builtin_huge_valf(); den[i] = 0.f; }
  if (i < NN * DD) agg[i] = 0.f;
}

__global__ void edge_logit(const float* __restrict__ qn, const float* __restrict__ ke,
                           const int* __restrict__ dst, float* __restrict__ logit,
                           float* __restrict__ mmax, int E) {
  int e = blockIdx.x * 256 + threadIdx.x;
  if (e >= E) return;
  int d0 = dst[e];
  const float* qr = qn + (size_t)d0 * DD;
  const float* kr = ke + (size_t)e * 96;
  float s = 0.f;
  for (int d = 0; d < DD; ++d) s += qr[d] * kr[d];
  s *= 0.10783277f;  // 1/sqrt(86)
  logit[e] = s;
  atomicMaxFloat(&mmax[d0], s);
}

__global__ void edge_a(const float* __restrict__ logit, const int* __restrict__ dst,
                       const float* __restrict__ mmax, float* __restrict__ aexp,
                       float* __restrict__ den, int E) {
  int e = blockIdx.x * 256 + threadIdx.x;
  if (e >= E) return;
  int d0 = dst[e];
  float mm = mmax[d0];
  if (mm < -1e37f) mm = 0.f;
  float a = expf(logit[e] - mm);
  aexp[e] = a;
  atomicAdd(&den[d0], a);
}

__global__ void edge_scatter(const float* __restrict__ aexp, const float* __restrict__ ve,
                             const int* __restrict__ dst, float* __restrict__ agg, int E) {
  int idx = blockIdx.x * 256 + threadIdx.x;
  if (idx >= E * DD) return;
  int e = idx / DD, d = idx - e * DD;
  atomicAdd(&agg[(size_t)dst[e] * DD + d], aexp[e] * ve[(size_t)e * 96 + d]);
}

// ---------------- fused node update: self + skip + MLP + LayerNorm ----------------

__global__ __launch_bounds__(192)
void node_update(const float* __restrict__ xin, int Fin,
                 const float* __restrict__ agg, const float* __restrict__ den,
                 const float* __restrict__ Wself, const float* __restrict__ W1,
                 const float* __restrict__ W2, int skip, float* __restrict__ out) {
  int n = blockIdx.x, tid = threadIdx.x;
  __shared__ float o0[DD];
  __shared__ float hh[DLIN];
  __shared__ float o1[DD];
  __shared__ float mred[2];
  const float* xr = xin + (size_t)n * Fin;
  float dinv = 1.f / (den[n] + 1e-9f);
  for (int d = tid; d < DD; d += 192) {
    float s = agg[(size_t)n * DD + d] * dinv;
    for (int f = 0; f < Fin; ++f) s += xr[f] * Wself[f * DD + d];
    if (skip) s += xr[d];
    o0[d] = s;
  }
  __syncthreads();
  for (int l = tid; l < DLIN; l += 192) {
    float s = 0.f;
    for (int d = 0; d < DD; ++d) s += o0[d] * W1[d * DLIN + l];
    hh[l] = gelu_tanh(s);
  }
  __syncthreads();
  for (int d = tid; d < DD; d += 192) {
    float s = o0[d];
    for (int l = 0; l < DLIN; ++l) s += hh[l] * W2[l * DD + d];
    o1[d] = s;
  }
  __syncthreads();
  if (tid == 0) {
    float mu = 0.f;
    for (int d = 0; d < DD; ++d) mu += o1[d];
    mu *= (1.f / DD);
    float var = 0.f;
    for (int d = 0; d < DD; ++d) { float t = o1[d] - mu; var += t * t; }
    var *= (1.f / DD);
    mred[0] = mu; mred[1] = rsqrtf(var + 1e-5f);
  }
  __syncthreads();
  for (int d = tid; d < DD; d += 192)
    out[(size_t)n * DD + d] = (o1[d] - mred[0]) * mred[1];
}

// ---------------- field evaluation: f[q,m] = sum_{n,s} shg[q,n,s] tmp[n,s,m] ----------------

__global__ __launch_bounds__(256)
void field_eval(const float* __restrict__ qp, const float* __restrict__ pos,
                const float* __restrict__ tmp, const float* __restrict__ Rs,
                float rc, int M, float* __restrict__ fout) {
  int q = blockIdx.x, tid = threadIdx.x;
  __shared__ float acc[96];
  __shared__ float coef[256 * 16];
  __shared__ float degS;
  for (int m = tid; m < M; m += 256) acc[m] = 0.f;
  if (tid == 0) degS = 0.f;
  float qx = qp[q * 3 + 0], qy = qp[q * 3 + 1], qz = qp[q * 3 + 2];
  float degL = 0.f;
  const float invw = 10.f / rc;       // 1/w, w = rc/NB
  const float mustep = rc * (1.f / 9.f);
  __syncthreads();
  for (int base = 0; base < NN; base += 256) {
    int n = base + tid;
    float c16[16];
    float dx = qx - pos[n * 3 + 0];
    float dy = qy - pos[n * 3 + 1];
    float dz = qz - pos[n * 3 + 2];
    float r = sqrtf(dx * dx + dy * dy + dz * dz + 1e-12f);
    bool ok = (r < rc) && (r > 1e-6f);
    if (ok) {
      float ir = 1.f / r;
      sphf(dx * ir, dy * ir, dz * ir, c16);
      float rb[NB_];
#pragma unroll
      for (int b = 0; b < NB_; ++b) {
        float zz = (r - (float)b * mustep) * invw;
        rb[b] = expf(-zz * zz);
      }
      float t = r / rc; t = t > 1.f ? 1.f : t;
      float env = 0.5f * (cosf(3.14159265f * t) + 1.f);
#pragma unroll
      for (int s = 0; s < 16; ++s) {
        float g = 0.f;
#pragma unroll
        for (int b = 0; b < NB_; ++b) g += rb[b] * Rs[b * 16 + s];
        c16[s] *= g * env;
      }
      degL += 1.f;
    } else {
#pragma unroll
      for (int s = 0; s < 16; ++s) c16[s] = 0.f;
    }
#pragma unroll
    for (int s = 0; s < 16; ++s) coef[tid * 16 + s] = c16[s];
    __syncthreads();
    for (int m = tid; m < M; m += 256) {
      float sum = acc[m];
      for (int j = 0; j < 256; ++j) {
        const float* tp = tmp + (size_t)(base + j) * (16 * M) + m;
        const float* cf = &coef[j * 16];
#pragma unroll
        for (int s = 0; s < 16; ++s) sum += cf[s] * tp[s * M];
      }
      acc[m] = sum;
    }
    __syncthreads();
  }
  atomicAdd(&degS, degL);
  __syncthreads();
  for (int m = tid; m < M; m += 256)
    fout[(size_t)q * M + m] = acc[m] * rsqrtf(degS + 1.f);
}

// ---------------- query position soft-argmax chain ----------------

__global__ void row_softmax_logw(const float* __restrict__ wl, float* __restrict__ logw) {
  int n = blockIdx.x * 256 + threadIdx.x;
  if (n >= NN) return;
  const float* r = wl + (size_t)n * 16;
  float mx = -1e38f;
  for (int j = 0; j < 16; ++j) mx = fmaxf(mx, r[j]);
  float s = 0.f;
  for (int j = 0; j < 16; ++j) s += expf(r[j] - mx);
  float inv = 1.f / s;
  for (int j = 0; j < 16; ++j)
    logw[(size_t)n * 16 + j] = logf(expf(r[j] - mx) * inv + 1e-4f) * (1.f / 0.03f);
}

__global__ __launch_bounds__(256)
void qpos_kernel(const float* __restrict__ logw, const float* __restrict__ pos,
                 float* __restrict__ qpos) {
  int q = blockIdx.x, tid = threadIdx.x;
  __shared__ float red[256];
  float mx = -1e38f;
  for (int n = tid; n < NN; n += 256) mx = fmaxf(mx, logw[(size_t)n * 16 + q]);
  red[tid] = mx; __syncthreads();
  for (int s = 128; s > 0; s >>= 1) { if (tid < s) red[tid] = fmaxf(red[tid], red[tid + s]); __syncthreads(); }
  float M = red[0]; __syncthreads();
  float sm = 0.f;
  for (int n = tid; n < NN; n += 256) sm += expf(logw[(size_t)n * 16 + q] - M);
  red[tid] = sm; __syncthreads();
  for (int s = 128; s > 0; s >>= 1) { if (tid < s) red[tid] += red[tid + s]; __syncthreads(); }
  float inv = 1.f / red[0]; __syncthreads();
  float ax = 0.f, ay = 0.f, az = 0.f;
  for (int n = tid; n < NN; n += 256) {
    float wgt = expf(logw[(size_t)n * 16 + q] - M) * inv;
    ax += wgt * pos[n * 3 + 0];
    ay += wgt * pos[n * 3 + 1];
    az += wgt * pos[n * 3 + 2];
  }
  red[tid] = ax; __syncthreads();
  for (int s = 128; s > 0; s >>= 1) { if (tid < s) red[tid] += red[tid + s]; __syncthreads(); }
  if (tid == 0) qpos[q * 3 + 0] = red[0];
  __syncthreads();
  red[tid] = ay; __syncthreads();
  for (int s = 128; s > 0; s >>= 1) { if (tid < s) red[tid] += red[tid + s]; __syncthreads(); }
  if (tid == 0) qpos[q * 3 + 1] = red[0];
  __syncthreads();
  red[tid] = az; __syncthreads();
  for (int s = 128; s > 0; s >>= 1) { if (tid < s) red[tid] += red[tid + s]; __syncthreads(); }
  if (tid == 0) qpos[q * 3 + 2] = red[0];
}

__global__ void qa_softmax(const float* __restrict__ raw, float* __restrict__ o) {
  float mx = -1e38f;
  for (int q = 0; q < 16; ++q) mx = fmaxf(mx, raw[q]);
  float s = 0.f;
  for (int q = 0; q < 16; ++q) s += expf(raw[q] - mx);
  float inv = 1.f / s;
  for (int q = 0; q < 16; ++q) o[q] = expf(raw[q] - mx) * inv;
}

// ---------------- host orchestration ----------------

extern "C" void kernel_launch(void* const* d_in, const int* in_sizes, int n_in,
                              void* d_out, int out_size, void* d_ws, size_t ws_size,
                              hipStream_t stream) {
  (void)n_in; (void)out_size; (void)ws_size;
  const float* pos = (const float*)d_in[0];
  const float* x0  = (const float*)d_in[1];
  const int* esrc  = (const int*)d_in[2];
  const int* edst  = (const int*)d_in[3];
  const int E = in_sizes[2];

  // pytree (sorted-key) param order: attn, feat, pos_w, se3t[7], se3t_feat[7]
  const float* attn_Rs = (const float*)d_in[4];
  const float* attn_W  = (const float*)d_in[5];
  const float* attn_Wl = (const float*)d_in[6];
  const float* attn_b  = (const float*)d_in[7];
  const float* feat_Rs = (const float*)d_in[8];
  const float* feat_W  = (const float*)d_in[9];
  const float* feat_Wl = (const float*)d_in[10];
  const float* feat_b  = (const float*)d_in[11];
  const float* posw_Rs = (const float*)d_in[12];
  const float* posw_W  = (const float*)d_in[13];
  const float* posw_Wl = (const float*)d_in[14];
  const float* posw_b  = (const float*)d_in[15];
  auto LP = [&](int base, int L, int w) -> const float* {
    return (const float*)d_in[base + L * 8 + w];   // Rk,Rv,W1,W2,Wk,Wq,Wself,Wv
  };

  char* wsb = (char*)d_ws;
  size_t off = 0;
  auto alloc = [&](size_t bytes) -> char* {
    char* p = wsb + off;
    off += (bytes + 255) & ~(size_t)255;
    return p;
  };
  float* esh   = (float*)alloc((size_t)E * 16 * 4);
  float* erad  = (float*)alloc((size_t)E * 10 * 4);
  float* bufA  = (float*)alloc((size_t)NN * DD * 4);
  float* bufB  = (float*)alloc((size_t)NN * DD * 4);
  float* qbuf  = (float*)alloc((size_t)NN * DD * 4);
  float* ke    = (float*)alloc((size_t)E * 96 * 4);
  float* ve    = (float*)alloc((size_t)E * 96 * 4);
  float* logit = (float*)alloc((size_t)E * 4);
  float* aexp  = (float*)alloc((size_t)E * 4);
  float* mmax  = (float*)alloc((size_t)NN * 4);
  float* den   = (float*)alloc((size_t)NN * 4);
  float* agg   = (float*)alloc((size_t)NN * DD * 4);
  __bf16* Wtk  = (__bf16*)alloc((size_t)96 * 1376 * 2);
  __bf16* Wtv  = (__bf16*)alloc((size_t)96 * 1376 * 2);
  float* tmpf  = (float*)alloc((size_t)NN * 16 * DD * 4);
  float* ffld  = (float*)alloc((size_t)NN * 32 * 4);
  float* wl    = (float*)alloc((size_t)NN * 16 * 4);
  float* logw  = (float*)alloc((size_t)NN * 16 * 4);
  float* qaraw = (float*)alloc(64 * 4);

  edge_geom<<<(E + 255) / 256, 256, 0, stream>>>(pos, esrc, edst, esh, erad, E);

  auto run_stack = [&](int pbase) -> const float* {
    const float* xin = x0;
    int Fin = 3;
    float* outBuf = bufA;
    for (int L = 0; L < 7; ++L) {
      const float* Rk = LP(pbase, L, 0);
      const float* Rv = LP(pbase, L, 1);
      const float* W1 = LP(pbase, L, 2);
      const float* W2 = LP(pbase, L, 3);
      const float* Wk = LP(pbase, L, 4);
      const float* Wq = LP(pbase, L, 5);
      const float* Ws = LP(pbase, L, 6);
      const float* Wv = LP(pbase, L, 7);
      const int Kpad = ((Fin * 16 + 31) / 32) * 32;
      const int cvg = (96 * Kpad + 255) / 256;
      conv_w_bf16<<<cvg, 256, 0, stream>>>(Wk, Wtk, Fin, Kpad);
      conv_w_bf16<<<cvg, 256, 0, stream>>>(Wv, Wtv, Fin, Kpad);
      gemm_rm<<<(NN * DD + 255) / 256, 256, 0, stream>>>(xin, Wq, nullptr, qbuf, NN, Fin, DD);
      dim3 gg((E + 63) / 64, 6);
      edge_gemm_wmma<<<gg, 128, 0, stream>>>(xin, Fin, esrc, esh, erad, Wtk, Kpad, Rk, ke, E);
      edge_gemm_wmma<<<gg, 128, 0, stream>>>(xin, Fin, esrc, esh, erad, Wtv, Kpad, Rv, ve, E);
      attn_init<<<(NN * DD + 255) / 256, 256, 0, stream>>>(mmax, den, agg);
      edge_logit<<<(E + 255) / 256, 256, 0, stream>>>(qbuf, ke, edst, logit, mmax, E);
      edge_a<<<(E + 255) / 256, 256, 0, stream>>>(logit, edst, mmax, aexp, den, E);
      edge_scatter<<<(E * DD + 255) / 256, 256, 0, stream>>>(aexp, ve, edst, agg, E);
      node_update<<<NN, 192, 0, stream>>>(xin, Fin, agg, den, Ws, W1, W2, (L > 0) ? 1 : 0, outBuf);
      xin = outBuf;
      Fin = DD;
      outBuf = (outBuf == bufA) ? bufB : bufA;
    }
    return xin;
  };

  // stack 1 -> feat1 -> query positions
  const float* feat1 = run_stack(16);
  gemm_rm<<<(NN * 16 * 32 + 255) / 256, 256, 0, stream>>>(feat1, posw_W, nullptr, tmpf, NN, DD, 16 * 32);
  field_eval<<<NN, 256, 0, stream>>>(pos, pos, tmpf, posw_Rs, FCUT, 32, ffld);
  gemm_rm<<<(NN * 16 + 255) / 256, 256, 0, stream>>>(ffld, posw_Wl, posw_b, wl, NN, 32, 16);
  row_softmax_logw<<<(NN + 255) / 256, 256, 0, stream>>>(wl, logw);
  float* qposO = (float*)d_out + 16 * DD;                     // qpos output region
  qpos_kernel<<<16, 256, 0, stream>>>(logw, pos, qposO);

  // stack 2 -> feat2 -> query fields
  const float* feat2 = run_stack(72);
  gemm_rm<<<(NN * 16 * DD + 255) / 256, 256, 0, stream>>>(feat2, feat_W, nullptr, tmpf, NN, DD, 16 * DD);
  field_eval<<<16, 256, 0, stream>>>(qposO, pos, tmpf, feat_Rs, FCUT, DD, ffld);
  gemm_rm<<<(16 * DD + 255) / 256, 256, 0, stream>>>(ffld, feat_Wl, feat_b, (float*)d_out, 16, DD, DD);
  gemm_rm<<<(NN * 16 * 10 + 255) / 256, 256, 0, stream>>>(feat2, attn_W, nullptr, tmpf, NN, DD, 16 * 10);
  field_eval<<<16, 256, 0, stream>>>(qposO, pos, tmpf, attn_Rs, FCUT, 10, ffld);
  gemm_rm<<<1, 256, 0, stream>>>(ffld, attn_Wl, attn_b, qaraw, 16, 10, 1);
  qa_softmax<<<1, 1, 0, stream>>>(qaraw, (float*)d_out + 16 * DD + 48);
}
